// SubBlock_88742614270606
// MI455X (gfx1250) — matrix operand order repaired and checked by reading, not compiled, not run on hardware
//
#include <hip/hip_runtime.h>

typedef _Float16 half_t;
typedef _Float16 v16h __attribute__((ext_vector_type(16)));
typedef float    v8f  __attribute__((ext_vector_type(8)));
typedef float    f4   __attribute__((ext_vector_type(4)));

union Frag16 { v16h h; f4 f[2]; };

// ---------------------------------------------------------------------------
// Kernel 1: pre = relu(bn1(x)), NCHW f32 -> NHWC f16 (LDS transpose tiles)
// x: [16,128,128,128], pre: [16,128,128,128] NHWC (b,h,w,ci)
// grid: (16, 2048)  block: 256
// ---------------------------------------------------------------------------
__global__ __launch_bounds__(256) void prep_pre(
    const float* __restrict__ x,
    const float* __restrict__ g, const float* __restrict__ bb,
    const float* __restrict__ mn, const float* __restrict__ vr,
    half_t* __restrict__ pre)
{
  __shared__ float tile[32][33];
  const int bh = blockIdx.y;          // b*128 + h
  const int b  = bh >> 7, h = bh & 127;
  const int wt = (blockIdx.x >> 2) * 32;  // w tile
  const int ct = (blockIdx.x & 3) * 32;   // ci tile
  const int tw = threadIdx.x & 31;
  const int tr = threadIdx.x >> 5;        // 0..7
#pragma unroll
  for (int p = 0; p < 4; ++p) {
    const int ci = ct + tr + p * 8;
    const float s  = g[ci] * rsqrtf(vr[ci] + 1e-5f);
    const float sh = bb[ci] - mn[ci] * s;
    float v = x[(((long)b * 128 + ci) * 128 + h) * 128 + wt + tw];
    v = v * s + sh;
    tile[tr + p * 8][tw] = v > 0.f ? v : 0.f;   // tile[ci_local][w_local]
  }
  __syncthreads();
#pragma unroll
  for (int p = 0; p < 4; ++p) {
    const int wl  = tr + p * 8;
    const int cil = tw;
    pre[(((long)(b * 128 + h) * 128) + wt + wl) * 128 + ct + cil] =
        (half_t)tile[cil][wl];
  }
}

// ---------------------------------------------------------------------------
// Weight repack: w[co][ci][3][3] f32 -> wt[co][(kh*3+kw)*CI + ci] f16
// ---------------------------------------------------------------------------
__global__ void prep_w3(const float* __restrict__ w, half_t* __restrict__ wt,
                        int CI, long tot)
{
  long i = (long)blockIdx.x * 256 + threadIdx.x;
  if (i >= tot) return;
  int kw = (int)(i % 3); long t = i / 3;
  int kh = (int)(t % 3); t /= 3;
  int ci = (int)(t % CI);
  int co = (int)(t / CI);
  wt[(long)co * (9L * CI) + (long)(kh * 3 + kw) * CI + ci] = (half_t)w[i];
}

__global__ void prep_w1x1(const float* __restrict__ w, half_t* __restrict__ wt, int n)
{
  int i = blockIdx.x * 256 + threadIdx.x;
  if (i < n) wt[i] = (half_t)w[i];   // [co][ci] flat is already [co][k]
}

// ---------------------------------------------------------------------------
// Implicit-GEMM conv via WMMA f32_16x16x32_f16.
//   M = B*64*64 = 65536 (b,ho,wo), N = 256 (co), K = KHW*KHW*CIN
//   A[m,k] = src NHWC f16 (bounds-checked im2col), Bw = [co][k] f16
// Block tile 128x128x32, 8 waves (4 along M x 2 along N), wave tile 32x64.
// EPI 0: h1 = (half)relu(bn2(acc))  -> outp as half_t [m][n]
// EPI 1: outp as float [m][n]
// grid: (512, 2)  block: 256
// ---------------------------------------------------------------------------
template<int CIN, int KHW, int STRIDE, int PADLO, int IH, int EPI>
__global__ __launch_bounds__(256, 2) void conv_gemm(
    const half_t* __restrict__ A, const half_t* __restrict__ Bw,
    void* __restrict__ outp,
    const float* __restrict__ g, const float* __restrict__ bb,
    const float* __restrict__ mn, const float* __restrict__ vr)
{
  constexpr int KTOT = KHW * KHW * CIN;
  constexpr int LDT  = 40;                 // padded LDS row stride (halves)
  __shared__ half_t lsA[2][128 * LDT];
  __shared__ half_t lsB[2][128 * LDT];

  const int tid = threadIdx.x;
  const int m0 = blockIdx.x * 128;
  const int n0 = blockIdx.y * 128;

  // ---- cooperative tile-load coordinates (256 threads, 32B each side) ----
  const int lr = tid >> 1;                 // row / col 0..127
  const int lc = (tid & 1) * 16;           // half-offset 0 or 16
  const int am  = m0 + lr;
  const int ab  = am >> 12;
  const int aho = (am >> 6) & 63;
  const int awo = am & 63;
  const long bbase = (long)(n0 + lr) * KTOT + lc;

  auto loadA = [&](int ks, half_t* dst) {
    const int f   = ks / CIN;
    const int kh  = f / KHW, kw = f % KHW;
    const int ci0 = ks % CIN;
    const int ih  = aho * STRIDE + kh - PADLO;
    const int iw  = awo * STRIDE + kw - PADLO;
    f4 v0 = {0.f, 0.f, 0.f, 0.f}, v1 = v0;
    if ((unsigned)ih < (unsigned)IH && (unsigned)iw < (unsigned)IH) {
      const half_t* p = A + (((long)(ab * IH + ih) * IH + iw) * CIN + ci0 + lc);
      v0 = *(const f4*)p;
      v1 = *(const f4*)(p + 8);
    }
    *(f4*)(dst + lr * LDT + lc)     = v0;
    *(f4*)(dst + lr * LDT + lc + 8) = v1;
  };
  auto loadB = [&](int ks, half_t* dst) {
    const half_t* p = Bw + bbase + ks;
    *(f4*)(dst + lr * LDT + lc)     = *(const f4*)p;
    *(f4*)(dst + lr * LDT + lc + 8) = *(const f4*)(p + 8);
  };

  // ---- wave / lane mapping ----
  const int lane  = tid & 31;
  const int wave  = tid >> 5;
  const int wm    = (wave & 3) * 32;       // wave M origin (0..96)
  const int wn    = (wave >> 2) * 64;      // wave N origin (0 or 64)
  const int lhalf = lane >> 4;             // 0: lanes 0-15, 1: lanes 16-31
  const int l15   = lane & 15;
  // 16-bit A layout: lanes 0-15 hold K 0-7 & 16-23; lanes 16-31 hold 8-15 & 24-31
  const int aco0  = lhalf ? 8 : 0;

  v8f acc[2][4];
  const v8f vzero = {0.f, 0.f, 0.f, 0.f, 0.f, 0.f, 0.f, 0.f};
#pragma unroll
  for (int fm = 0; fm < 2; ++fm)
#pragma unroll
    for (int fn = 0; fn < 4; ++fn) acc[fm][fn] = vzero;

  loadA(0, lsA[0]);
  loadB(0, lsB[0]);
  int buf = 0;
  for (int ks = 0; ks < KTOT; ks += 32) {
    __syncthreads();
    if (ks + 32 < KTOT) {                  // prefetch next tile
      loadA(ks + 32, lsA[buf ^ 1]);
      loadB(ks + 32, lsB[buf ^ 1]);
    }
    Frag16 fa[2], fb[4];
#pragma unroll
    for (int fm = 0; fm < 2; ++fm) {
      const half_t* pr = &lsA[buf][(wm + fm * 16 + l15) * LDT];
      fa[fm].f[0] = *(const f4*)(pr + aco0);
      fa[fm].f[1] = *(const f4*)(pr + aco0 + 16);
    }
#pragma unroll
    for (int fn = 0; fn < 4; ++fn) {
      // B layout: lane holds column n, K 0-7 & 16-23 (lanes 0-15) / 8-15 & 24-31 (lanes 16-31)
      const half_t* pc = &lsB[buf][(wn + fn * 16 + l15) * LDT + lhalf * 8];
      fb[fn].f[0] = *(const f4*)pc;
      fb[fn].f[1] = *(const f4*)(pc + 16);
    }
#pragma unroll
    for (int fm = 0; fm < 2; ++fm)
#pragma unroll
      for (int fn = 0; fn < 4; ++fn)
        acc[fm][fn] = __builtin_amdgcn_wmma_f32_16x16x32_f16(
            false, fa[fm].h, false, fb[fn].h, (short)0, acc[fm][fn],
            false, false);
    buf ^= 1;
  }

  // ---- epilogue ----
#pragma unroll
  for (int fn = 0; fn < 4; ++fn) {
    const int n = n0 + wn + fn * 16 + l15;
    float scale = 1.f, shift = 0.f;
    if constexpr (EPI == 0) {
      scale = g[n] * rsqrtf(vr[n] + 1e-5f);
      shift = bb[n] - mn[n] * scale;
    }
#pragma unroll
    for (int fm = 0; fm < 2; ++fm) {
#pragma unroll
      for (int r = 0; r < 8; ++r) {
        const int row = wm + fm * 16 + (lhalf ? 8 + r : r);
        const long m  = m0 + row;
        float v = acc[fm][fn][r];
        if constexpr (EPI == 0) {
          v = v * scale + shift;
          v = v > 0.f ? v : 0.f;
          ((half_t*)outp)[m * 256 + n] = (half_t)v;
        } else {
          ((float*)outp)[m * 256 + n] = v;
        }
      }
    }
  }
}

// ---------------------------------------------------------------------------
// SE global average pool: s[b][co] = mean_{hw} h2[(b*4096+hw)*256 + co]
// grid: 64 (b*4 + co_block), block: 256
// ---------------------------------------------------------------------------
__global__ __launch_bounds__(256) void se_pool(const float* __restrict__ h2,
                                               float* __restrict__ s)
{
  __shared__ float red[4][64];
  const int b    = blockIdx.x >> 2;
  const int cb   = (blockIdx.x & 3) * 64;
  const int co   = cb + (threadIdx.x & 63);
  const int part = threadIdx.x >> 6;
  float sum = 0.f;
  for (int hw = part; hw < 4096; hw += 4)
    sum += h2[((long)(b << 12) + hw) * 256 + co];
  red[part][threadIdx.x & 63] = sum;
  __syncthreads();
  if (part == 0) {
    const int t = threadIdx.x;  // 0..63
    sum = red[0][t] + red[1][t] + red[2][t] + red[3][t];
    s[b * 256 + cb + t] = sum * (1.f / 4096.f);
  }
}

// ---------------------------------------------------------------------------
// SE FCs: a = sigmoid(relu(s @ w1^T + b1) @ w2^T + b2)  [16,256]
// one block, 256 threads
// ---------------------------------------------------------------------------
__global__ __launch_bounds__(256) void se_fc(
    const float* __restrict__ s,
    const float* __restrict__ w1, const float* __restrict__ b1,
    const float* __restrict__ w2, const float* __restrict__ b2,
    float* __restrict__ a)
{
  __shared__ float sl[16 * 256];
  __shared__ float hid[16][16];
  for (int i = threadIdx.x; i < 16 * 256; i += 256) sl[i] = s[i];
  __syncthreads();
  {
    const int bi = threadIdx.x >> 4, j = threadIdx.x & 15;
    float acc = b1[j];
    for (int ci = 0; ci < 256; ++ci) acc += sl[bi * 256 + ci] * w1[j * 256 + ci];
    hid[bi][j] = acc > 0.f ? acc : 0.f;
  }
  __syncthreads();
#pragma unroll
  for (int t = 0; t < 16; ++t) {
    const int o  = threadIdx.x + 256 * t;
    const int bi = o >> 8, co = o & 255;
    float acc = b2[co];
#pragma unroll
    for (int j = 0; j < 16; ++j) acc += hid[bi][j] * w2[co * 16 + j];
    a[o] = 1.f / (1.f + __expf(-acc));
  }
}

// ---------------------------------------------------------------------------
// Final: out(NCHW) = h2[m][n] * a[b][n] + res[m][n], LDS-transposed for
// coalesced reads (along n) and writes (along wo).
// grid: (4, 1024) block: 256
// ---------------------------------------------------------------------------
__global__ __launch_bounds__(256) void epilogue_se(
    const float* __restrict__ h2, const float* __restrict__ res,
    const float* __restrict__ a, float* __restrict__ out)
{
  __shared__ float t[64][65];
  const int bho = blockIdx.y;             // b*64 + ho
  const int b = bho >> 6, ho = bho & 63;
  const int n0 = blockIdx.x * 64;
  const long m0 = (long)bho * 64;         // covers wo = 0..63
  const int nl = threadIdx.x & 63;
  const int r0 = threadIdx.x >> 6;
  const float an = a[b * 256 + n0 + nl];
#pragma unroll
  for (int p = 0; p < 16; ++p) {
    const int r = r0 + p * 4;             // wo
    const long idx = (m0 + r) * 256 + n0 + nl;
    t[nl][r] = h2[idx] * an + res[idx];
  }
  __syncthreads();
  const int wl = threadIdx.x & 63;
  const int nr = threadIdx.x >> 6;
#pragma unroll
  for (int p = 0; p < 16; ++p) {
    const int n = nr + p * 4;
    out[(((long)b * 256 + n0 + n) * 64 + ho) * 64 + wl] = t[n][wl];
  }
}

// ---------------------------------------------------------------------------
extern "C" void kernel_launch(void* const* d_in, const int* in_sizes, int n_in,
                              void* d_out, int out_size, void* d_ws, size_t ws_size,
                              hipStream_t stream)
{
  const float* x      = (const float*)d_in[0];
  const float* bn1_g  = (const float*)d_in[1];
  const float* bn1_b  = (const float*)d_in[2];
  const float* bn1_m  = (const float*)d_in[3];
  const float* bn1_v  = (const float*)d_in[4];
  const float* bn2_g  = (const float*)d_in[5];
  const float* bn2_b  = (const float*)d_in[6];
  const float* bn2_m  = (const float*)d_in[7];
  const float* bn2_v  = (const float*)d_in[8];
  const float* w_conv1= (const float*)d_in[9];
  const float* w_conv2= (const float*)d_in[10];
  const float* w_res  = (const float*)d_in[11];
  const float* se_w1  = (const float*)d_in[12];
  const float* se_b1  = (const float*)d_in[13];
  const float* se_w2  = (const float*)d_in[14];
  const float* se_b2  = (const float*)d_in[15];

  char* ws = (char*)d_ws;
  half_t* pre = (half_t*)(ws);                      //  67108864 B
  half_t* h1  = (half_t*)(ws + 67108864L);          //  33554432 B
  float*  h2  = (float*) (ws + 100663296L);         //  67108864 B
  float*  res = (float*) (ws + 167772160L);         //  67108864 B
  half_t* w1t = (half_t*)(ws + 234881024L);         //    589824 B
  half_t* w2t = (half_t*)(ws + 235470848L);         //   1179648 B
  half_t* wrt = (half_t*)(ws + 236650496L);         //     65536 B
  float*  sse = (float*) (ws + 236716032L);         //     16384 B
  float*  ase = (float*) (ws + 236732416L);         //     16384 B

  // stage 0: activation transform + weight repacks (independent)
  prep_pre<<<dim3(16, 2048), 256, 0, stream>>>(x, bn1_g, bn1_b, bn1_m, bn1_v, pre);
  prep_w3<<<(294912 + 255) / 256, 256, 0, stream>>>(w_conv1, w1t, 128, 294912L);
  prep_w3<<<(589824 + 255) / 256, 256, 0, stream>>>(w_conv2, w2t, 256, 589824L);
  prep_w1x1<<<128, 256, 0, stream>>>(w_res, wrt, 32768);

  // stage 1: conv1 (3x3 s2, bn2+relu -> f16) and residual 1x1 s2 conv (f32)
  conv_gemm<128, 3, 2, 0, 128, 0><<<dim3(512, 2), 256, 0, stream>>>(
      pre, w1t, h1, bn2_g, bn2_b, bn2_m, bn2_v);
  conv_gemm<128, 1, 2, 0, 128, 1><<<dim3(512, 2), 256, 0, stream>>>(
      pre, wrt, res, nullptr, nullptr, nullptr, nullptr);

  // stage 2: conv2 (3x3 s1 -> f32)
  conv_gemm<256, 3, 1, 1, 64, 1><<<dim3(512, 2), 256, 0, stream>>>(
      h1, w2t, h2, nullptr, nullptr, nullptr, nullptr);

  // stage 3: SE attention
  se_pool<<<64, 256, 0, stream>>>(h2, sse);
  se_fc<<<1, 256, 0, stream>>>(sse, se_w1, se_b1, se_w2, se_b2, ase);

  // stage 4: scale + residual add, write NCHW output
  epilogue_se<<<dim3(4, 1024), 256, 0, stream>>>(h2, res, ase, (float*)d_out);
}
